// DialogueRnn_57183194579288
// MI455X (gfx1250) — matrix operand order repaired and checked by reading, not compiled, not run on hardware
//
#include <hip/hip_runtime.h>
#include <hip/hip_bf16.h>

// ---------------- problem constants ----------------
constexpr int Tn = 256;   // timesteps (q axis too)
constexpr int Bn = 128;   // batch
constexpr int Pn = 9;     // parties
constexpr int DtC = 600, DvC = 300, DaC = 300;
constexpr int Dd = 512;   // Du=Dg=Dp=De=Dh
constexpr int G3 = 1536;  // 3*512
constexpr int Cn = 7;
constexpr int Kf  = 1200; // Dt+Dv+Da
constexpr int Kfp = 1216; // padded to multiple of 32

typedef __bf16 bf16_t;
typedef __attribute__((ext_vector_type(16))) __bf16 v16bf;
typedef __attribute__((ext_vector_type(8)))  __bf16 v8bf;
typedef __attribute__((ext_vector_type(8)))  float  v8f;
typedef unsigned int u32x4 __attribute__((ext_vector_type(4)));
typedef int          i32x8 __attribute__((ext_vector_type(8)));
typedef int          i32x4 __attribute__((ext_vector_type(4)));

static __device__ __forceinline__ bf16_t f2bf(float f) {
    union { float f; unsigned u; } v; v.f = f;
    unsigned r = (v.u + 0x7FFFu + ((v.u >> 16) & 1u)) >> 16;
    unsigned short h = (unsigned short)r;
    bf16_t out;
    __builtin_memcpy(&out, &h, 2);
    return out;
}
static __device__ __forceinline__ float sigmoidf_(float x) { return 1.0f / (1.0f + expf(-x)); }

static __device__ __forceinline__ v16bf ldfragA(const bf16_t* ap) {
    // A fragment: elems 0..7 -> K = 8*half + i ; elems 8..15 -> K = 16 + 8*half + i
    v8bf lo = *(const v8bf*)(ap);
    v8bf hi = *(const v8bf*)(ap + 16);
    return __builtin_shufflevector(lo, hi, 0,1,2,3,4,5,6,7,8,9,10,11,12,13,14,15);
}
static __device__ __forceinline__ v16bf ldfragB(const bf16_t* bp) {
    // B fragment: elems 0..7 -> K = 8*half + i ; elems 8..15 -> K = 16 + 8*half + i
    v8bf lo = *(const v8bf*)(bp);
    v8bf hi = *(const v8bf*)(bp + 16);
    return __builtin_shufflevector(lo, hi, 0,1,2,3,4,5,6,7,8,9,10,11,12,13,14,15);
}
static __device__ __forceinline__ v8f wmma_bf16(v16bf a, v16bf b, v8f c) {
    return __builtin_amdgcn_wmma_f32_16x16x32_bf16(false, a, false, b, (short)0, c, false, false);
}

// ---- TDM: load a 2D tile (rows x 32 bf16) from global (row stride ld elems) to LDS ----
// D# per CDNA5 ISA ch.8: group0 {count=1 | lds_addr | global_addr | type=2},
// group1 {data_size=2B, tensor_dim0=32, tensor_dim1=rows, tile_dim0=32, tile_dim1=rows,
//         tensor_dim0_stride=ld}. 2D tile: groups 2/3 zero.
// This toolchain's builtin is the 6-arg form:
//   (uint32x4 g0, int32x8 g1, int32x4 g2, int32x4 g3, int32x8 extra, int cpol)
static __device__ __forceinline__ void tdm_load_2d(unsigned lds_off,
                                                   unsigned long long gaddr,
                                                   unsigned rows,
                                                   unsigned long long ld_elems)
{
    u32x4 g0;
    g0[0] = 1u;                                        // [1:0] count=1, user desc
    g0[1] = lds_off;                                   // [63:32] LDS byte address
    g0[2] = (unsigned)(gaddr & 0xFFFFFFFFu);           // global_addr[31:0]
    g0[3] = (unsigned)((gaddr >> 32) & 0x01FFFFFFu)    // global_addr[56:32]
            | 0x80000000u;                             // [127:126] type=2 ("image")
    i32x8 g1;
    g1[0] = 0x00010000;                                // [17:16] data_size=1 (2 bytes)
    g1[1] = (int)(32u << 16);                          // [63:48] tensor_dim0 = 32
    g1[2] = (int)(rows << 16);                         // [95:80] tensor_dim1 = rows
    g1[3] = (int)(32u << 16);                          // [127:112] tile_dim0 = 32
    g1[4] = (int)rows;                                 // [143:128] tile_dim1 = rows
    g1[5] = (int)(unsigned)(ld_elems & 0xFFFFFFFFull); // [191:160] dim0_stride lo
    g1[6] = (int)(unsigned)((ld_elems >> 32) & 0xFFFFull); // [207:192] dim0_stride hi
    g1[7] = 0;
    i32x4 gz  = {0, 0, 0, 0};
    i32x8 gz8 = {0, 0, 0, 0, 0, 0, 0, 0};
    __builtin_amdgcn_tensor_load_to_lds(g0, g1, gz, gz, gz8, 0);
}

// ---------------- v1: per-wave 16x16 WMMA GEMM (scan-sized problems) ----------------
// C[b, m, n] = sum_k A[b, m, k] * Bw[b, n, k] + bias[n]
__global__ void gemm_wmma_bf16(const bf16_t* __restrict__ A, long lda, long sA,
                               const bf16_t* __restrict__ Bw, long ldb, long sB,
                               const float* __restrict__ bias,
                               float* __restrict__ Cf, bf16_t* __restrict__ Cbf,
                               long ldc, long sC,
                               int M, int N, int K, int batch, int act)
{
    int wave = (int)((blockIdx.x * (long)blockDim.x + threadIdx.x) >> 5);
    int lane = threadIdx.x & 31;
    int tilesN = N >> 4;
    int tilesPerBatch = (M >> 4) * tilesN;
    int total = tilesPerBatch * batch;
    if (wave >= total) return;              // wave-uniform: EXEC stays all-ones for WMMA
    int b  = wave / tilesPerBatch;
    int t  = wave % tilesPerBatch;
    int tm = t / tilesN, tn = t % tilesN;

    int half = lane >> 4;
    int r    = lane & 15;

    const bf16_t* arow = A  + (long)b * sA + (long)(tm * 16 + r) * lda;
    const bf16_t* brow = Bw + (long)b * sB + (long)(tn * 16 + r) * ldb;

    v8f acc = {};
    for (int k = 0; k < K; k += 32) {
        v16bf av = ldfragA(arow + k + 8 * half);
        v16bf bv = ldfragB(brow + k + 8 * half);
        acc = wmma_bf16(av, bv, acc);
    }
    int n = tn * 16 + r;
    float bv = bias ? bias[n] : 0.0f;
    #pragma unroll
    for (int i = 0; i < 8; ++i) {
        int m = tm * 16 + half * 8 + i;
        float v = acc[i] + bv;
        if (act == 1) v = fmaxf(v, 0.0f);
        long idx = (long)b * sC + (long)m * ldc + n;
        if (Cf)  Cf[idx]  = v;
        if (Cbf) Cbf[idx] = f2bf(v);
    }
}

// ---------------- v2: TDM + LDS double-buffered GEMM (large problems) ----------------
// Block = 256 threads (8 waves) computes a 64x128 C tile; each wave a 32x32 patch
// (2x2 WMMA tiles). K consumed in 32-wide panels staged to LDS by the Tensor Data
// Mover (wave 0 issues; TENSORcnt is per-wave & in-order so wait(2) retires the
// previous panel pair). Requires M%64==0, N%128==0, K%32==0.
__global__ void __launch_bounds__(256) gemm_wmma_lds(
    const bf16_t* __restrict__ A, long lda, long sA,
    const bf16_t* __restrict__ Bw, long ldb, long sB,
    const float* __restrict__ bias,
    float* __restrict__ Cf, bf16_t* __restrict__ Cbf,
    long ldc, long sC,
    int M, int N, int K, int batch, int act)
{
    __shared__ __align__(16) bf16_t As[2][64][32];    // 8 KB
    __shared__ __align__(16) bf16_t Bs[2][128][32];   // 16 KB

    int blocksN = N >> 7;
    int blocksPerBatch = (M >> 6) * blocksN;
    int bidx = blockIdx.x;
    int b  = bidx / blocksPerBatch;
    int t  = bidx % blocksPerBatch;
    int bm = (t / blocksN) << 6;
    int bn = (t % blocksN) << 7;

    int tid  = threadIdx.x;
    int wid  = tid >> 5;
    int lane = tid & 31;
    int wm = (wid >> 2) << 5;             // 0 / 32
    int wn = (wid & 3) << 5;              // 0 / 32 / 64 / 96
    int half = lane >> 4, r = lane & 15;

    const bf16_t* Ab = A  + (long)b * sA + (long)bm * lda;
    const bf16_t* Bb = Bw + (long)b * sB + (long)bn * ldb;

    int P = K >> 5;
    v8f acc00 = {}, acc01 = {}, acc10 = {}, acc11 = {};

    if (wid == 0) {
        tdm_load_2d((unsigned)(size_t)&As[0][0][0], (unsigned long long)(size_t)Ab,
                    64u, (unsigned long long)lda);
        tdm_load_2d((unsigned)(size_t)&Bs[0][0][0], (unsigned long long)(size_t)Bb,
                    128u, (unsigned long long)ldb);
    }
    for (int p = 0; p < P; ++p) {
        int buf = p & 1;
        if (wid == 0) {
            if (p + 1 < P) {
                long koff = (long)(p + 1) << 5;
                tdm_load_2d((unsigned)(size_t)&As[buf ^ 1][0][0],
                            (unsigned long long)(size_t)(Ab + koff),
                            64u, (unsigned long long)lda);
                tdm_load_2d((unsigned)(size_t)&Bs[buf ^ 1][0][0],
                            (unsigned long long)(size_t)(Bb + koff),
                            128u, (unsigned long long)ldb);
                __builtin_amdgcn_s_wait_tensorcnt(2);   // panel p's pair retired
            } else {
                __builtin_amdgcn_s_wait_tensorcnt(0);
            }
        }
        __syncthreads();
        v16bf a0 = ldfragA(&As[buf][wm + r][8 * half]);
        v16bf a1 = ldfragA(&As[buf][wm + 16 + r][8 * half]);
        v16bf b0 = ldfragB(&Bs[buf][wn + r][8 * half]);
        v16bf b1 = ldfragB(&Bs[buf][wn + 16 + r][8 * half]);
        acc00 = wmma_bf16(a0, b0, acc00);
        acc01 = wmma_bf16(a0, b1, acc01);
        acc10 = wmma_bf16(a1, b0, acc10);
        acc11 = wmma_bf16(a1, b1, acc11);
        __syncthreads();   // all reads of buf done before wave0 refills it next iter
    }
    // epilogue: 4 tiles; acc[i] = D[m0 + 8*half + i][n0 + r]
    #pragma unroll
    for (int mt = 0; mt < 2; ++mt) {
        #pragma unroll
        for (int nt = 0; nt < 2; ++nt) {
            v8f acc = (mt == 0) ? (nt == 0 ? acc00 : acc01)
                                : (nt == 0 ? acc10 : acc11);
            int n = bn + wn + nt * 16 + r;
            float bv = bias ? bias[n] : 0.0f;
            #pragma unroll
            for (int i = 0; i < 8; ++i) {
                int m = bm + wm + mt * 16 + half * 8 + i;
                float v = acc[i] + bv;
                if (act == 1) v = fmaxf(v, 0.0f);
                long idx = (long)b * sC + (long)m * ldc + n;
                if (Cf)  Cf[idx]  = v;
                if (Cbf) Cbf[idx] = f2bf(v);
            }
        }
    }
}

// ---------------- setup kernels ----------------
__global__ void cvt_pad_k(const float* __restrict__ src, bf16_t* __restrict__ dst,
                          int rows, int cols, int dcols)
{
    long j = blockIdx.x * (long)blockDim.x + threadIdx.x;
    long n = (long)rows * dcols;
    if (j >= n) return;
    int rr = (int)(j / dcols), c = (int)(j % dcols);
    dst[j] = (c < cols) ? f2bf(src[(long)rr * cols + c]) : f2bf(0.0f);
}

__global__ void build_X_k(const float* __restrict__ text, const float* __restrict__ video,
                          const float* __restrict__ audio, bf16_t* __restrict__ X)
{
    long j = blockIdx.x * (long)blockDim.x + threadIdx.x;
    long n = (long)Tn * Bn * Kfp;
    if (j >= n) return;
    int row = (int)(j / Kfp), c = (int)(j % Kfp);
    float v;
    if      (c < DtC)       v = text [(long)row * DtC + c];
    else if (c < DtC + DvC) v = video[(long)row * DvC + (c - DtC)];
    else if (c < Kf)        v = audio[(long)row * DaC + (c - DtC - DvC)];
    else                    v = 0.0f;
    X[j] = f2bf(v);
}

__global__ void spk_argmax_k(const float* __restrict__ pm, int* __restrict__ spk)
{
    int j = blockIdx.x * blockDim.x + threadIdx.x;
    if (j >= Tn * Bn) return;
    const float* p = pm + (long)j * Pn;
    int best = 0; float bv = p[0];
    for (int q = 1; q < Pn; ++q) { float v = p[q]; if (v > bv) { bv = v; best = q; } }
    spk[j] = best;
}

__global__ void fill_f32_k(float* p, float v, int n)
{
    int j = blockIdx.x * blockDim.x + threadIdx.x;
    if (j < n) p[j] = v;
}

// ---------------- scan kernels ----------------
__global__ void step_pre_k(const bf16_t* __restrict__ u_t, const float* __restrict__ personal,
                           const int* __restrict__ spk_t, const float* __restrict__ acc,
                           const float* __restrict__ lsum, int tstep,
                           bf16_t* __restrict__ Ag, bf16_t* __restrict__ Ap)
{
    int j = blockIdx.x * blockDim.x + threadIdx.x;
    if (j >= Bn * Dd) return;
    int b = j / Dd, d = j % Dd;
    bf16_t u = u_t[j];
    Ag[(long)b * 2 * Dd + d] = u;
    Ap[(long)b * 2 * Dd + d] = u;
    int s = spk_t[b];
    float sp = personal[((long)b * Pn + s) * Dd + d];
    Ag[(long)b * 2 * Dd + Dd + d] = f2bf(sp);
    float c = 0.0f;
    if (tstep > 0) c = acc[j] / fmaxf(lsum[b], 1e-30f);
    Ap[(long)b * 2 * Dd + Dd + d] = f2bf(c);
}

__global__ void gru_point_k(const float* __restrict__ gi, const float* __restrict__ gh,
                            const float* __restrict__ h, float* __restrict__ out,
                            bf16_t* __restrict__ out_bf, float* __restrict__ out2,
                            bf16_t* __restrict__ out2_bf)
{
    int j = blockIdx.x * blockDim.x + threadIdx.x;
    if (j >= Bn * Dd) return;
    int b = j / Dd, d = j % Dd;
    long base = (long)b * G3;
    float r = sigmoidf_(gi[base + d]          + gh[base + d]);
    float z = sigmoidf_(gi[base + Dd + d]     + gh[base + Dd + d]);
    float n = tanhf    (gi[base + 2 * Dd + d] + r * gh[base + 2 * Dd + d]);
    float v = (1.0f - z) * n + z * h[j];
    out[j] = v;
    if (out_bf)  out_bf[j]  = f2bf(v);
    if (out2)    out2[j]    = v;
    if (out2_bf) out2_bf[j] = f2bf(v);
}

__global__ void gru_personal_k(const float* __restrict__ gi, const float* __restrict__ gh,
                               float* __restrict__ personal, const int* __restrict__ spk_t,
                               bf16_t* __restrict__ sp_bf)
{
    int j = blockIdx.x * blockDim.x + threadIdx.x;
    if (j >= Bn * Dd) return;
    int b = j / Dd, d = j % Dd;
    int s = spk_t[b];
    long hidx = ((long)b * Pn + s) * Dd + d;
    long base = (long)b * G3;
    float h = personal[hidx];
    float r = sigmoidf_(gi[base + d]          + gh[base + d]);
    float z = sigmoidf_(gi[base + Dd + d]     + gh[base + Dd + d]);
    float n = tanhf    (gi[base + 2 * Dd + d] + r * gh[base + 2 * Dd + d]);
    float v = (1.0f - z) * n + z * h;
    personal[hidx] = v;
    sp_bf[j] = f2bf(v);
}

__global__ void attn_update_k(const float* __restrict__ g, const float* __restrict__ w,
                              float* __restrict__ m, float* __restrict__ l,
                              float* __restrict__ acc)
{
    int b = blockIdx.x, tid = threadIdx.x;
    __shared__ float red[256];
    __shared__ float sh_scale, sh_e;
    float s = 0.0f;
    for (int d = tid; d < Dd; d += 256) s += g[(long)b * Dd + d] * w[d];
    red[tid] = s; __syncthreads();
    for (int off = 128; off; off >>= 1) { if (tid < off) red[tid] += red[tid + off]; __syncthreads(); }
    if (tid == 0) {
        float sv = red[0];
        float mo = m[b];
        float mn = fmaxf(mo, sv);
        float scale = expf(mo - mn);
        float e = expf(sv - mn);
        m[b] = mn;
        l[b] = l[b] * scale + e;
        sh_scale = scale; sh_e = e;
    }
    __syncthreads();
    float scale = sh_scale, e = sh_e;
    for (int d = tid; d < Dd; d += 256) {
        long i = (long)b * Dd + d;
        acc[i] = acc[i] * scale + e * g[i];
    }
}

// ---------------- epilogue kernels ----------------
__global__ void mmask_build_k(const float* __restrict__ emotions, const float* __restrict__ mask,
                              bf16_t* __restrict__ out)
{
    long j = blockIdx.x * (long)blockDim.x + threadIdx.x;
    long n = (long)Tn * Bn * Dd;
    if (j >= n) return;
    int t = (int)(j / ((long)Bn * Dd));
    int b = (int)((j / Dd) % Bn);
    out[j] = f2bf(emotions[j] * mask[(long)b * Tn + t]);
}

__global__ void emoT_build_k(const float* __restrict__ emotions, bf16_t* __restrict__ emoT)
{
    long j = blockIdx.x * (long)blockDim.x + threadIdx.x;
    long n = (long)Bn * Dd * Tn;
    if (j >= n) return;
    int b = (int)(j / ((long)Dd * Tn));
    int rr = (int)(j % ((long)Dd * Tn));
    int d = rr / Tn, t = rr % Tn;
    emoT[j] = f2bf(emotions[((long)t * Bn + b) * Dd + d]);
}

__global__ void attn_softmax_k(const float* __restrict__ al, const float* __restrict__ mask,
                               bf16_t* __restrict__ out)
{
    int qb = blockIdx.x;
    int b = qb % Bn;
    int t = threadIdx.x;
    __shared__ float red[256];
    float mk = mask[(long)b * Tn + t];
    float v = tanhf(al[(long)qb * Tn + t] * mk);
    red[t] = v; __syncthreads();
    for (int off = 128; off; off >>= 1) { if (t < off) red[t] = fmaxf(red[t], red[t + off]); __syncthreads(); }
    float mx = red[0]; __syncthreads();
    float e = expf(v - mx) * mk;
    red[t] = e; __syncthreads();
    for (int off = 128; off; off >>= 1) { if (t < off) red[t] += red[t + off]; __syncthreads(); }
    float s = red[0];
    out[(long)qb * Tn + t] = f2bf(e / s);
}

__global__ void final_logits_k(const float* __restrict__ hidden, const float* __restrict__ Ws,
                               const float* __restrict__ bs, float* __restrict__ out)
{
    int row = blockIdx.x * blockDim.x + threadIdx.x;
    if (row >= Tn * Bn) return;
    const float* h = hidden + (long)row * Dd;
    float lg[Cn];
    for (int c = 0; c < Cn; ++c) {
        float s = bs[c];
        const float* w = Ws + (long)c * Dd;
        for (int d = 0; d < Dd; ++d) s += h[d] * w[d];
        lg[c] = s;
    }
    float mx = lg[0];
    for (int c = 1; c < Cn; ++c) mx = fmaxf(mx, lg[c]);
    float se = 0.0f;
    for (int c = 0; c < Cn; ++c) se += expf(lg[c] - mx);
    float lse = mx + logf(se);
    for (int c = 0; c < Cn; ++c) out[(long)row * Cn + c] = lg[c] - lse;
}

// ---------------- host ----------------
#define CDIV(a, b) (((a) + (b) - 1) / (b))

extern "C" void kernel_launch(void* const* d_in, const int* in_sizes, int n_in,
                              void* d_out, int out_size, void* d_ws, size_t ws_size,
                              hipStream_t stream)
{
    const float* text   = (const float*)d_in[0];
    const float* video  = (const float*)d_in[1];
    const float* audio  = (const float*)d_in[2];
    const float* pmask  = (const float*)d_in[3];
    const float* mask   = (const float*)d_in[4];
    const float* Wf     = (const float*)d_in[5];
    const float* bf_b   = (const float*)d_in[6];
    const float* Wgi    = (const float*)d_in[7];
    const float* Wgh    = (const float*)d_in[8];
    const float* bgi    = (const float*)d_in[9];
    const float* bgh    = (const float*)d_in[10];
    const float* Wpi    = (const float*)d_in[11];
    const float* Wph    = (const float*)d_in[12];
    const float* bpi    = (const float*)d_in[13];
    const float* bph    = (const float*)d_in[14];
    const float* Wei    = (const float*)d_in[15];
    const float* Weh    = (const float*)d_in[16];
    const float* bei    = (const float*)d_in[17];
    const float* beh    = (const float*)d_in[18];
    const float* w_attn = (const float*)d_in[19];
    const float* Wm     = (const float*)d_in[20];
    const float* bm     = (const float*)d_in[21];
    const float* Wl     = (const float*)d_in[22];
    const float* bl     = (const float*)d_in[23];
    const float* Ws     = (const float*)d_in[24];
    const float* bs     = (const float*)d_in[25];
    float* out = (float*)d_out;

    char* ws = (char*)d_ws;
    const size_t MBy = 1ull << 20;
    bf16_t* utter_bf = (bf16_t*)(ws + 0);            // 32MB
    bf16_t* Xbf      = (bf16_t*)(ws + 32 * MBy);     // 80MB, dead after utter GEMM
    bf16_t* x_bf     = (bf16_t*)(ws + 32 * MBy);     // 32MB (reuse)
    bf16_t* Mmask_bf = (bf16_t*)(ws + 64 * MBy);     // 32MB (reuse)
    float*  emotions = (float*) (ws + 112 * MBy);    // 64MB
    bf16_t* emo_all_bf = (bf16_t*)(ws + 176 * MBy);  // 32MB
    float*  al_f     = (float*) (ws + 208 * MBy);    // 32MB
    bf16_t* al_bf    = (bf16_t*)(ws + 240 * MBy);    // 16MB
    bf16_t* emoT_bf  = (bf16_t*)(ws + 256 * MBy);    // 32MB
    bf16_t* pooled_bf= (bf16_t*)(ws + 288 * MBy);    // 32MB
    float*  hidden_f = (float*) (ws + 320 * MBy);    // 64MB
    char* wofs = ws + 384 * MBy;
    bf16_t* Wf_bf  = (bf16_t*)(wofs);                    wofs += (size_t)Dd * Kfp * 2;
    bf16_t* Wgi_bf = (bf16_t*)(wofs);                    wofs += (size_t)G3 * 1024 * 2;
    bf16_t* Wgh_bf = (bf16_t*)(wofs);                    wofs += (size_t)G3 * Dd * 2;
    bf16_t* Wpi_bf = (bf16_t*)(wofs);                    wofs += (size_t)G3 * 1024 * 2;
    bf16_t* Wph_bf = (bf16_t*)(wofs);                    wofs += (size_t)G3 * Dd * 2;
    bf16_t* Wei_bf = (bf16_t*)(wofs);                    wofs += (size_t)G3 * Dd * 2;
    bf16_t* Weh_bf = (bf16_t*)(wofs);                    wofs += (size_t)G3 * Dd * 2;
    bf16_t* Wm_bf  = (bf16_t*)(wofs);                    wofs += (size_t)Dd * Dd * 2;
    bf16_t* Wl_bf  = (bf16_t*)(wofs);                    wofs += (size_t)Dd * Dd * 2;
    char* sofs = ws + 400 * MBy;
    int*    spk      = (int*)   (sofs);  sofs += (size_t)Tn * Bn * 4;
    bf16_t* Ag       = (bf16_t*)(sofs);  sofs += (size_t)Bn * 1024 * 2;
    bf16_t* Ap       = (bf16_t*)(sofs);  sofs += (size_t)Bn * 1024 * 2;
    float*  giBuf    = (float*) (sofs);  sofs += (size_t)Bn * G3 * 4;
    float*  ghBuf    = (float*) (sofs);  sofs += (size_t)Bn * G3 * 4;
    float*  g_f      = (float*) (sofs);  sofs += (size_t)Bn * Dd * 4;
    bf16_t* g_bf     = (bf16_t*)(sofs);  sofs += (size_t)Bn * Dd * 2;
    float*  emo_f    = (float*) (sofs);  sofs += (size_t)Bn * Dd * 4;
    bf16_t* emo_bf   = (bf16_t*)(sofs);  sofs += (size_t)Bn * Dd * 2;
    bf16_t* sp_bf    = (bf16_t*)(sofs);  sofs += (size_t)Bn * Dd * 2;
    float*  m_f      = (float*) (sofs);  sofs += 4096;
    float*  l_f      = (float*) (sofs);  sofs += 4096;
    float*  acc_f    = (float*) (sofs);  sofs += (size_t)Bn * Dd * 4;
    float*  personal = (float*) (sofs);  sofs += (size_t)Bn * Pn * Dd * 4;

    auto gemm = [&](const bf16_t* A, long lda, long sA,
                    const bf16_t* Bw, long ldb, long sB,
                    const float* bias, float* Cf, bf16_t* Cbf, long ldc, long sC,
                    int M, int N, int K, int batch, int act) {
        long waves = (long)batch * (M / 16) * (N / 16);
        int blocks = (int)CDIV(waves * 32, 256);
        gemm_wmma_bf16<<<blocks, 256, 0, stream>>>(A, lda, sA, Bw, ldb, sB, bias,
                                                   Cf, Cbf, ldc, sC, M, N, K, batch, act);
    };
    auto gemm2 = [&](const bf16_t* A, long lda, long sA,
                     const bf16_t* Bw, long ldb, long sB,
                     const float* bias, float* Cf, bf16_t* Cbf, long ldc, long sC,
                     int M, int N, int K, int batch, int act) {
        int blocks = batch * (M / 64) * (N / 128);
        gemm_wmma_lds<<<blocks, 256, 0, stream>>>(A, lda, sA, Bw, ldb, sB, bias,
                                                  Cf, Cbf, ldc, sC, M, N, K, batch, act);
    };
    auto cvt = [&](const float* src, bf16_t* dst, int rows, int cols, int dcols) {
        long n = (long)rows * dcols;
        cvt_pad_k<<<(int)CDIV(n, 256), 256, 0, stream>>>(src, dst, rows, cols, dcols);
    };

    // -------- phase 0: operand prep --------
    cvt(Wf,  Wf_bf,  Dd, Kf,  Kfp);
    cvt(Wgi, Wgi_bf, G3, 1024, 1024);
    cvt(Wgh, Wgh_bf, G3, Dd,  Dd);
    cvt(Wpi, Wpi_bf, G3, 1024, 1024);
    cvt(Wph, Wph_bf, G3, Dd,  Dd);
    cvt(Wei, Wei_bf, G3, Dd,  Dd);
    cvt(Weh, Weh_bf, G3, Dd,  Dd);
    cvt(Wm,  Wm_bf,  Dd, Dd,  Dd);
    cvt(Wl,  Wl_bf,  Dd, Dd,  Dd);
    {
        long n = (long)Tn * Bn * Kfp;
        build_X_k<<<(int)CDIV(n, 256), 256, 0, stream>>>(text, video, audio, Xbf);
    }
    spk_argmax_k<<<CDIV(Tn * Bn, 256), 256, 0, stream>>>(pmask, spk);

    fill_f32_k<<<1, 256, 0, stream>>>(m_f, -1.0e9f, Bn);
    (void)hipMemsetAsync(l_f, 0, Bn * 4, stream);
    (void)hipMemsetAsync(acc_f, 0, (size_t)Bn * Dd * 4, stream);
    (void)hipMemsetAsync(g_f, 0, (size_t)Bn * Dd * 4, stream);
    (void)hipMemsetAsync(g_bf, 0, (size_t)Bn * Dd * 2, stream);
    (void)hipMemsetAsync(emo_f, 0, (size_t)Bn * Dd * 4, stream);
    (void)hipMemsetAsync(emo_bf, 0, (size_t)Bn * Dd * 2, stream);
    (void)hipMemsetAsync(personal, 0, (size_t)Bn * Pn * Dd * 4, stream);

    // -------- phase 1: utter = X @ Wf^T + bf (TDM-staged GEMM) --------
    gemm2(Xbf, Kfp, 0, Wf_bf, Kfp, 0, bf_b, nullptr, utter_bf, Dd, 0,
          Tn * Bn, Dd, Kfp, 1, 0);

    // -------- phase 2: sequential scan (latency-bound; per-wave GEMM) --------
    const int BD = Bn * Dd;
    const int bdBlocks = CDIV(BD, 256);
    for (int t = 0; t < Tn; ++t) {
        const bf16_t* u_t   = utter_bf + (size_t)t * BD;
        const int*    spk_t = spk + (size_t)t * Bn;
        step_pre_k<<<bdBlocks, 256, 0, stream>>>(u_t, personal, spk_t, acc_f, l_f, t, Ag, Ap);
        gemm(Ag,   1024, 0, Wgi_bf, 1024, 0, bgi, giBuf, nullptr, G3, 0, Bn, G3, 1024, 1, 0);
        gemm(g_bf, Dd,   0, Wgh_bf, Dd,   0, bgh, ghBuf, nullptr, G3, 0, Bn, G3, Dd,   1, 0);
        gru_point_k<<<bdBlocks, 256, 0, stream>>>(giBuf, ghBuf, g_f, g_f, g_bf, nullptr, nullptr);
        attn_update_k<<<Bn, 256, 0, stream>>>(g_f, w_attn, m_f, l_f, acc_f);
        gemm(Ap,       1024, 0, Wpi_bf, 1024, 0, bpi, giBuf, nullptr, G3, 0, Bn, G3, 1024, 1, 0);
        gemm(Ag + Dd,  1024, 0, Wph_bf, Dd,   0, bph, ghBuf, nullptr, G3, 0, Bn, G3, Dd,   1, 0);
        gru_personal_k<<<bdBlocks, 256, 0, stream>>>(giBuf, ghBuf, personal, spk_t, sp_bf);
        gemm(sp_bf,  Dd, 0, Wei_bf, Dd, 0, bei, giBuf, nullptr, G3, 0, Bn, G3, Dd, 1, 0);
        gemm(emo_bf, Dd, 0, Weh_bf, Dd, 0, beh, ghBuf, nullptr, G3, 0, Bn, G3, Dd, 1, 0);
        gru_point_k<<<bdBlocks, 256, 0, stream>>>(giBuf, ghBuf, emo_f, emo_f, emo_bf,
                                                  emotions + (size_t)t * BD,
                                                  emo_all_bf + (size_t)t * BD);
    }

    // -------- phase 3: attention pooling epilogue (TDM-staged GEMMs) --------
    long nTBD = (long)Tn * Bn * Dd;
    gemm2(emo_all_bf, Dd, 0, Wm_bf, Dd, 0, bm, nullptr, x_bf, Dd, 0, Tn * Bn, Dd, Dd, 1, 0);
    mmask_build_k<<<(int)CDIV(nTBD, 256), 256, 0, stream>>>(emotions, mask, Mmask_bf);
    gemm2(x_bf, (long)Bn * Dd, Dd, Mmask_bf, (long)Bn * Dd, Dd, nullptr,
          al_f, nullptr, (long)Bn * Tn, Tn, Tn, Tn, Dd, Bn, 0);
    attn_softmax_k<<<Tn * Bn, Tn, 0, stream>>>(al_f, mask, al_bf);
    emoT_build_k<<<(int)CDIV(nTBD, 256), 256, 0, stream>>>(emotions, emoT_bf);
    gemm2(al_bf, (long)Bn * Tn, Tn, emoT_bf, Tn, (long)Dd * Tn, nullptr,
          nullptr, pooled_bf, (long)Bn * Dd, Dd, Tn, Dd, Tn, Bn, 0);
    gemm2(pooled_bf, Dd, 0, Wl_bf, Dd, 0, bl, hidden_f, nullptr, Dd, 0,
          Tn * Bn, Dd, Dd, 1, 1);
    final_logits_k<<<CDIV(Tn * Bn, 256), 256, 0, stream>>>(hidden_f, Ws, bs, out);
}
